// NeuralSplineFlow_83640193123053
// MI455X (gfx1250) — matrix operand order repaired and checked by reading, not compiled, not run on hardware
//
#include <hip/hip_runtime.h>
#include <math.h>

typedef __attribute__((ext_vector_type(16))) __bf16 v16bf;
typedef __attribute__((ext_vector_type(8)))  float  v8f;

#define N_T       8
#define NDIM      16
#define NCTX      32
#define NIN       48
#define KIN       64          // K padded for GEMM1
#define HID       128
#define NP        97
#define NBINS     32
#define BATCH     131072
#define MIN_BIN   1.0e-4f
#define MIN_SLOPE 1.0e-4f
#define SM_SCALE  (1.0f - NBINS * MIN_BIN)
#define LRELU     0.01f
#define LOG2PI    1.8378770664093453f

// per-sample param row layout (f32)
#define SLP0      64          // slopes (33) processed in place
#define X0        100         // xpos (33 cumulative widths)
#define Y0        134         // ypos (33 cumulative heights)
#define PPAD      168

// packed bf16 A-fragment tiles in d_ws (each tile: 32 lanes x 16 halves = 1KB)
#define W1_TILES  (N_T * 8 * 2)          // 128
#define W2_TILES  (N_T * 8 * 4)          // 256
#define W3_TILES  (N_T * 7 * 4)          // 224
#define TOT_TILES (W1_TILES + W2_TILES + W3_TILES)
#define W2_BASE   W1_TILES
#define W3_BASE   (W1_TILES + W2_TILES)

#define WMMA_BF16(a, b, c) \
  __builtin_amdgcn_wmma_f32_16x16x32_bf16(false, (a), false, (b), (short)0, (c), false, false)

// ---------------------------------------------------------------------------
// Pack f32 weights into bf16 WMMA A-fragments (16x32 tile, contiguous k order, same order as the B reads):
// lane l: half = l>>4, M = mtile*16 + (l&15)
// v16bf element e: j = e>>1, b = e&1,
//   k = ktile*32 + half*16 + e
// ---------------------------------------------------------------------------
__global__ void nsf_pack(const float* __restrict__ W1,
                         const float* __restrict__ W2,
                         const float* __restrict__ W3,
                         __bf16* __restrict__ wp) {
  int gid  = blockIdx.x * blockDim.x + threadIdx.x;
  int tile = gid >> 5;
  int lane = gid & 31;
  if (tile >= TOT_TILES) return;

  const float* W; int t, mt, kt, Kv, Mv;
  if (tile < W1_TILES) {
    int r = tile; t = r >> 4; r &= 15; mt = r >> 1; kt = r & 1;
    W = W1 + (size_t)t * HID * NIN; Kv = NIN; Mv = HID;
  } else if (tile < W3_BASE) {
    int r = tile - W2_BASE; t = r >> 5; r &= 31; mt = r >> 2; kt = r & 3;
    W = W2 + (size_t)t * HID * HID; Kv = HID; Mv = HID;
  } else {
    int r = tile - W3_BASE; t = r / 28; r %= 28; mt = r >> 2; kt = r & 3;
    W = W3 + (size_t)t * NP * HID; Kv = HID; Mv = NP;
  }

  int half = lane >> 4, ml = lane & 15;
  int m = mt * 16 + ml;
  __bf16* dst = wp + (size_t)tile * 512 + lane * 16;
#pragma unroll
  for (int e = 0; e < 16; ++e) {
    int j = e >> 1, b = e & 1;
    int k = kt * 32 + half * 16 + e; (void)j; (void)b;
    float v = (m < Mv && k < Kv) ? W[(size_t)m * Kv + k] : 0.0f;
    dst[e] = (__bf16)v;
  }
}

// ---------------------------------------------------------------------------
// Main kernel: one wave per block owning 32 samples (two WMMA N-tiles), so
// every A-fragment load feeds two WMMAs. Wave-private LDS => no barriers.
// ---------------------------------------------------------------------------
__global__ __launch_bounds__(32) void nsf_main(
    const float* __restrict__ x,   const float* __restrict__ ctx,
    const float* __restrict__ b1,  const float* __restrict__ b2,
    const float* __restrict__ b3,  const __bf16* __restrict__ wp,
    float* __restrict__ out) {

  __shared__ __align__(32) __bf16 sh_h[32 * KIN];   // net input  [n][K]
  __shared__ __align__(32) __bf16 sh_A[32 * HID];   // H1         [n][128]
  __shared__ __align__(32) __bf16 sh_B[32 * HID];   // H2         [n][128]
  __shared__ float sh_p[32 * PPAD];                 // params + xpos/ypos
  __shared__ float sh_y[32 * NDIM];                 // flow state

  const int lane  = threadIdx.x & 31;
  const int half  = lane >> 4;
  const int n16   = lane & 15;
  const int sbase = blockIdx.x * 32;

  const v16bf* wpf = (const v16bf*)wp;

  // init flow state y = x  (16 elements per lane)
  for (int i = lane; i < 32 * NDIM; i += 32)
    sh_y[i] = x[(size_t)(sbase + (i >> 4)) * NDIM + (i & 15)];

  // static part of h: context at K=16..47, zero pad K=48..63
  for (int i = lane; i < 32 * KIN; i += 32) {
    int nn = i >> 6, k = i & 63;
    float v = (k >= 16 && k < 48) ? ctx[(size_t)(sbase + nn) * NCTX + (k - 16)]
                                  : 0.0f;
    sh_h[i] = (__bf16)v;
  }

  float ld_acc = 0.0f;
  const float sp_off = logf(expm1f(1.0f - MIN_SLOPE));

  for (int t = N_T - 1; t >= 0; --t) {
    const int passBit = (t & 1) ? 0 : 1;  // dims with (d&1)==passBit pass thru

    // refresh masked x-part of h (K=0..15)
    for (int i = lane; i < 32 * NDIM; i += 32) {
      int nn = i >> 4, d = i & 15;
      float v = ((d & 1) == passBit) ? sh_y[nn * NDIM + d] : 0.0f;
      sh_h[nn * KIN + d] = (__bf16)v;
    }

    // ---- GEMM1: W1 (128 x 64pad) * h (64 x 32) -> H1, leaky relu ----
    for (int mt = 0; mt < 8; ++mt) {
      v8f acc0, acc1;
#pragma unroll
      for (int v = 0; v < 8; ++v) {
        float bv = b1[t * HID + mt * 16 + v + 8 * half];
        acc0[v] = bv; acc1[v] = bv;
      }
#pragma unroll
      for (int kt = 0; kt < 2; ++kt) {
        v16bf a   = wpf[(size_t)(t * 16 + mt * 2 + kt) * 32 + lane];
        v16bf bb0 = *(const v16bf*)&sh_h[n16 * KIN + kt * 32 + half * 16];
        v16bf bb1 = *(const v16bf*)&sh_h[(16 + n16) * KIN + kt * 32 + half * 16];
        acc0 = WMMA_BF16(a, bb0, acc0);
        acc1 = WMMA_BF16(a, bb1, acc1);
      }
#pragma unroll
      for (int v = 0; v < 8; ++v) {
        float f0 = acc0[v]; f0 = (f0 > 0.0f) ? f0 : LRELU * f0;
        float f1 = acc1[v]; f1 = (f1 > 0.0f) ? f1 : LRELU * f1;
        sh_A[n16 * HID + mt * 16 + v + 8 * half]        = (__bf16)f0;
        sh_A[(16 + n16) * HID + mt * 16 + v + 8 * half] = (__bf16)f1;
      }
    }

    // ---- GEMM2: W2 (128 x 128) * H1 -> H2, leaky relu ----
    for (int mt = 0; mt < 8; ++mt) {
      v8f acc0, acc1;
#pragma unroll
      for (int v = 0; v < 8; ++v) {
        float bv = b2[t * HID + mt * 16 + v + 8 * half];
        acc0[v] = bv; acc1[v] = bv;
      }
#pragma unroll
      for (int kt = 0; kt < 4; ++kt) {
        v16bf a   = wpf[(size_t)(W2_BASE + t * 32 + mt * 4 + kt) * 32 + lane];
        v16bf bb0 = *(const v16bf*)&sh_A[n16 * HID + kt * 32 + half * 16];
        v16bf bb1 = *(const v16bf*)&sh_A[(16 + n16) * HID + kt * 32 + half * 16];
        acc0 = WMMA_BF16(a, bb0, acc0);
        acc1 = WMMA_BF16(a, bb1, acc1);
      }
#pragma unroll
      for (int v = 0; v < 8; ++v) {
        float f0 = acc0[v]; f0 = (f0 > 0.0f) ? f0 : LRELU * f0;
        float f1 = acc1[v]; f1 = (f1 > 0.0f) ? f1 : LRELU * f1;
        sh_B[n16 * HID + mt * 16 + v + 8 * half]        = (__bf16)f0;
        sh_B[(16 + n16) * HID + mt * 16 + v + 8 * half] = (__bf16)f1;
      }
    }

    // ---- GEMM3: W3 (97->112pad x 128) * H2 -> params (f32) ----
    for (int mt = 0; mt < 7; ++mt) {
      v8f acc0, acc1;
#pragma unroll
      for (int v = 0; v < 8; ++v) {
        int M = mt * 16 + v + 8 * half;
        float bv = (M < NP) ? b3[t * NP + M] : 0.0f;
        acc0[v] = bv; acc1[v] = bv;
      }
#pragma unroll
      for (int kt = 0; kt < 4; ++kt) {
        v16bf a   = wpf[(size_t)(W3_BASE + t * 28 + mt * 4 + kt) * 32 + lane];
        v16bf bb0 = *(const v16bf*)&sh_B[n16 * HID + kt * 32 + half * 16];
        v16bf bb1 = *(const v16bf*)&sh_B[(16 + n16) * HID + kt * 32 + half * 16];
        acc0 = WMMA_BF16(a, bb0, acc0);
        acc1 = WMMA_BF16(a, bb1, acc1);
      }
#pragma unroll
      for (int v = 0; v < 8; ++v) {
        int M = mt * 16 + v + 8 * half;
        sh_p[n16 * PPAD + M]        = acc0[v];
        sh_p[(16 + n16) * PPAD + M] = acc1[v];
      }
    }

    // ---- RQ spline (f32). One lane per sample. ----
    float* pp = &sh_p[lane * PPAD];

    {   // widths softmax -> cumulative xpos
      float mx = pp[0];
      for (int k = 1; k < NBINS; ++k) mx = fmaxf(mx, pp[k]);
      float s = 0.0f;
      for (int k = 0; k < NBINS; ++k) s += __expf(pp[k] - mx);
      float inv = SM_SCALE / s;
      float cum = 0.0f;
      pp[X0] = 0.0f;
      for (int k = 0; k < NBINS; ++k) {
        cum += __expf(pp[k] - mx) * inv + MIN_BIN;
        pp[X0 + 1 + k] = cum;
      }
    }
    {   // heights softmax -> cumulative ypos
      float mx = pp[NBINS];
      for (int k = 1; k < NBINS; ++k) mx = fmaxf(mx, pp[NBINS + k]);
      float s = 0.0f;
      for (int k = 0; k < NBINS; ++k) s += __expf(pp[NBINS + k] - mx);
      float inv = SM_SCALE / s;
      float cum = 0.0f;
      pp[Y0] = 0.0f;
      for (int k = 0; k < NBINS; ++k) {
        cum += __expf(pp[NBINS + k] - mx) * inv + MIN_BIN;
        pp[Y0 + 1 + k] = cum;
      }
    }
    {   // softplus slopes (33 values), in place
      for (int k = 0; k <= NBINS; ++k) {
        float v  = pp[SLP0 + k] + sp_off;
        float sp = (v > 20.0f) ? v : log1pf(__expf(v));
        pp[SLP0 + k] = sp + MIN_SLOPE;
      }
    }

    // transform the 8 non-masked dims of this lane's sample
    const int tp = 1 - passBit;   // parity of transformed dims
    for (int jj = 0; jj < 8; ++jj) {
      int d = tp + 2 * jj;
      float xv = sh_y[lane * NDIM + d];
      bool inside = (xv >= 0.0f) && (xv <= 1.0f);

      // idx = clip(searchsorted(xpos, xv, right)-1, 0, 31)
      int idx = 0;
      for (int k = 1; k < NBINS; ++k)
        if (xv >= pp[X0 + k]) idx = k;

      float xk  = pp[X0 + idx];
      float wk  = pp[X0 + idx + 1] - xk;
      float yk  = pp[Y0 + idx];
      float hk  = pp[Y0 + idx + 1] - yk;
      float dk  = pp[SLP0 + idx];
      float dk1 = pp[SLP0 + idx + 1];

      float sk  = hk / wk;
      float z   = (xv - xk) / wk;
      z = fminf(fmaxf(z, 0.0f), 1.0f);
      float omz = 1.0f - z;
      float z1  = z * omz;
      float den = sk + (dk1 + dk - 2.0f * sk) * z1;
      float yv  = yk + hk * (sk * z * z + dk * z1) / den;
      float ld  = 2.0f * __logf(sk)
                + __logf(dk1 * z * z + 2.0f * sk * z1 + dk * omz * omz)
                - 2.0f * __logf(den);

      if (inside) {
        sh_y[lane * NDIM + d] = yv;
        ld_acc += ld;
      }
    }
  }

  // base log prob + accumulated log-det (one full sample per lane)
  float ss = 0.0f;
  for (int j = 0; j < NDIM; ++j) {
    float v = sh_y[lane * NDIM + j];
    ss += v * v;
  }
  out[sbase + lane] = -0.5f * ss - 0.5f * NDIM * LOG2PI + ld_acc;
}

// ---------------------------------------------------------------------------
extern "C" void kernel_launch(void* const* d_in, const int* in_sizes, int n_in,
                              void* d_out, int out_size, void* d_ws, size_t ws_size,
                              hipStream_t stream) {
  const float* x   = (const float*)d_in[0];
  const float* ctx = (const float*)d_in[1];
  const float* W1  = (const float*)d_in[2];
  const float* b1  = (const float*)d_in[3];
  const float* W2  = (const float*)d_in[4];
  const float* b2  = (const float*)d_in[5];
  const float* W3  = (const float*)d_in[6];
  const float* b3  = (const float*)d_in[7];
  __bf16* wp = (__bf16*)d_ws;   // needs 608 KiB of workspace

  int packThreads = TOT_TILES * 32;
  nsf_pack<<<(packThreads + 255) / 256, 256, 0, stream>>>(W1, W2, W3, wp);

  int nblocks = BATCH / 32;     // 1 wave x 32 samples per block
  nsf_main<<<nblocks, 32, 0, stream>>>(x, ctx, b1, b2, b3, wp, (float*)d_out);
}
